// SelfAttentionLayer_9672266350973
// MI455X (gfx1250) — compile-verified, matrix operand order checked by reading, not run on hardware
//
#include <hip/hip_runtime.h>

#define NN 8192
#define HH 1024
#define OO 1024

typedef __bf16 bf16_t;
typedef __attribute__((ext_vector_type(16))) __bf16 bf16x16;
typedef __attribute__((ext_vector_type(8)))  float  floatx8;

union FragU { bf16x16 v; float4 q[2]; };

static __device__ __forceinline__ floatx8 wmma_bf16(bf16x16 a, bf16x16 b, floatx8 c) {
  // D = A(16x32 bf16) x B(32x16 bf16) + C(16x16 f32)
  return __builtin_amdgcn_wmma_f32_16x16x32_bf16(false, a, false, b, (short)0, c, false, false);
}

// Load a 16x32 bf16 A/B fragment from a row-major matrix.
// A-layout (wave32): lanes 0-15 row M=lane, K=[0..7]+[16..23]; lanes 16-31 same rows, K=[8..15]+[24..31].
// B uses the same per-lane pattern with N<->M, i.e. it reads rows of B^T — so for X @ W^T we feed rows of W.
static __device__ __forceinline__ bf16x16 load_frag(const bf16_t* __restrict__ base,
                                                    int row0, int ld, int k0) {
  const int lane = threadIdx.x & 31;
  const bf16_t* p = base + (size_t)(row0 + (lane & 15)) * ld + (k0 + ((lane >> 4) << 3));
  FragU u;
  u.q[0] = *reinterpret_cast<const float4*>(p);        // K = kh+0..7
  u.q[1] = *reinterpret_cast<const float4*>(p + 16);   // K = 16+kh+0..7
  return u.v;
}

// ---------------------------------------------------------------- convert
__global__ __launch_bounds__(256) void f32_to_bf16_kernel(const float* __restrict__ in,
                                                          bf16_t* __restrict__ out, int n4) {
  int i = blockIdx.x * blockDim.x + threadIdx.x;
  if (i >= n4) return;
  float4 v = reinterpret_cast<const float4*>(in)[i];
  union { bf16_t h[4]; float2 f2; } u;
  u.h[0] = (bf16_t)v.x; u.h[1] = (bf16_t)v.y; u.h[2] = (bf16_t)v.z; u.h[3] = (bf16_t)v.w;
  reinterpret_cast<float2*>(out)[i] = u.f2;
}

// ---------------------------------------------------------------- fused QKV projection
// grid (N/128, O/128, 3): z selects Q/K/V. 8 waves per block: 4 row-groups x 2 col-groups,
// each wave computes a 32x64 tile (2x4 WMMA accumulators). V is stored transposed (VT [O,N]).
__global__ __launch_bounds__(256) void qkv_kernel(
    const bf16_t* __restrict__ Xb,
    const bf16_t* __restrict__ Wqb, const bf16_t* __restrict__ Wkb, const bf16_t* __restrict__ Wvb,
    const float* __restrict__ bq, const float* __restrict__ bk, const float* __restrict__ bv,
    bf16_t* __restrict__ Qb, bf16_t* __restrict__ Kb, bf16_t* __restrict__ VTb)
{
  const int z = blockIdx.z;
  const bf16_t* __restrict__ W = (z == 0) ? Wqb : (z == 1) ? Wkb : Wvb;
  const float* __restrict__ bias = (z == 0) ? bq : (z == 1) ? bk : bv;

  const int w = threadIdx.x >> 5;
  const int lane = threadIdx.x & 31;
  const int rowBase = blockIdx.x * 128 + (w >> 1) * 32;
  const int colBase = blockIdx.y * 128 + (w & 1) * 64;

  const floatx8 fzero = {0.f, 0.f, 0.f, 0.f, 0.f, 0.f, 0.f, 0.f};
  floatx8 acc[2][4];
#pragma unroll
  for (int i = 0; i < 2; ++i)
#pragma unroll
    for (int j = 0; j < 4; ++j) acc[i][j] = fzero;

  for (int k0 = 0; k0 < HH; k0 += 32) {
    bf16x16 a0 = load_frag(Xb, rowBase,      HH, k0);
    bf16x16 a1 = load_frag(Xb, rowBase + 16, HH, k0);
    bf16x16 b0 = load_frag(W,  colBase +  0, HH, k0);
    bf16x16 b1 = load_frag(W,  colBase + 16, HH, k0);
    bf16x16 b2 = load_frag(W,  colBase + 32, HH, k0);
    bf16x16 b3 = load_frag(W,  colBase + 48, HH, k0);
    acc[0][0] = wmma_bf16(a0, b0, acc[0][0]);
    acc[0][1] = wmma_bf16(a0, b1, acc[0][1]);
    acc[0][2] = wmma_bf16(a0, b2, acc[0][2]);
    acc[0][3] = wmma_bf16(a0, b3, acc[0][3]);
    acc[1][0] = wmma_bf16(a1, b0, acc[1][0]);
    acc[1][1] = wmma_bf16(a1, b1, acc[1][1]);
    acc[1][2] = wmma_bf16(a1, b2, acc[1][2]);
    acc[1][3] = wmma_bf16(a1, b3, acc[1][3]);
  }

  // C layout: lane holds column n = lane&15; VGPR e holds row e + 8*(lane>>4).
  const int cq = lane & 15;
  const int rh = (lane >> 4) << 3;
#pragma unroll
  for (int j = 0; j < 4; ++j) {
    const int col = colBase + j * 16 + cq;
    const float bb = bias[col];
#pragma unroll
    for (int i = 0; i < 2; ++i) {
      const int r0 = rowBase + i * 16 + rh;
      if (z == 2) {
        // store transposed: VT[col][r0..r0+7] is contiguous -> one 16B store
        union { bf16_t h[8]; float4 q; } u;
#pragma unroll
        for (int e = 0; e < 8; ++e) u.h[e] = (bf16_t)(acc[i][j][e] + bb);
        *reinterpret_cast<float4*>(VTb + (size_t)col * NN + r0) = u.q;
      } else {
        bf16_t* __restrict__ Out = (z == 0) ? Qb : Kb;
#pragma unroll
        for (int e = 0; e < 8; ++e)
          Out[(size_t)(r0 + e) * OO + col] = (bf16_t)(acc[i][j][e] + bb);
      }
    }
  }
}

// ---------------------------------------------------------------- flash attention
// grid (N/32), 256 threads (8 waves). Workgroup owns 32 queries.
// Per 64-key iteration:
//   phase 1: wave w computes S^T tile (kt = w>>1 over 4 key sub-tiles, qt = w&1 over 2 query tiles),
//            S^T = K @ Q^T so softmax reduction is in-lane + one shfl_xor(16).
//   online softmax stats combined deterministically via small LDS arrays by threads 0..31.
//   P (bf16) published to LDS as ready-made B fragments (lane-aligned, zero shuffles).
//   phase 2: wave w accumulates O^T[d-slice 128][32 q] += V^T @ P^T with 16 WMMA accumulators.
__global__ __launch_bounds__(256) void attn_kernel(
    const bf16_t* __restrict__ Qb, const bf16_t* __restrict__ Kb,
    const bf16_t* __restrict__ VTb, float* __restrict__ out)
{
  __shared__ float m_state[32], l_state[32], m_new_s[32], cfac_s[32];
  __shared__ float tmax[4][32], tsum[4][32];
  __shared__ float4 pfragq[2][2][32][2];   // [kpair][qtile][lane][half16B] = P^T B-fragments

  const int tid = threadIdx.x;
  const int w = tid >> 5;
  const int lane = tid & 31;
  const int qBase = blockIdx.x * 32;
  const int kt = w >> 1, qt = w & 1;
  const int kp = kt >> 1, khalf = kt & 1;
  const int d0 = w * 128;                  // private d-slice for phase 2
  const int cq = lane & 15;

  if (tid < 32) { m_state[tid] = -3.0e38f; l_state[tid] = 0.f; }

  const floatx8 fzero = {0.f, 0.f, 0.f, 0.f, 0.f, 0.f, 0.f, 0.f};
  floatx8 oacc[8][2];
#pragma unroll
  for (int dt = 0; dt < 8; ++dt) { oacc[dt][0] = fzero; oacc[dt][1] = fzero; }

  const float sscale = 1.44269504f / 32.0f;   // log2(e)/sqrt(O): softmax in base-2 domain

  for (int j0 = 0; j0 < NN; j0 += 64) {
    // prefetch the NEXT key-block working set while this block computes
    // (emits global_prefetch_b8; every enabled lane pulls one cacheline)
    if (j0 + 64 < NN) {
      // V^T: next block is VT[d0..d0+127][j0+64..j0+127] -> 4 rows per lane covers the slice
#pragma unroll
      for (int r = 0; r < 4; ++r) {
        const bf16_t* pv = VTb + (size_t)(d0 + r * 32 + lane) * NN + (j0 + 64);
        __builtin_prefetch(pv, 0, 3);
      }
      // K: warm the first line of each of the next 64 key rows (waves 0-1 cover all rows)
      if (w < 2) {
        const bf16_t* pk = Kb + (size_t)(j0 + 64 + w * 32 + lane) * HH;
        __builtin_prefetch(pk, 0, 3);
      }
    }

    // ---- phase 1: S^T tile (16 keys x 16 queries), f32 accum
    floatx8 s = fzero;
    for (int h0 = 0; h0 < HH; h0 += 32) {
      bf16x16 a = load_frag(Kb, j0 + kt * 16,  HH, h0);   // A = K rows
      bf16x16 b = load_frag(Qb, qBase + qt * 16, HH, h0); // B = Q^T (rows of Q)
      s = wmma_bf16(a, b, s);
    }
    float smax = -3.0e38f;
#pragma unroll
    for (int e = 0; e < 8; ++e) { s[e] *= sscale; smax = fmaxf(smax, s[e]); }
    smax = fmaxf(smax, __shfl_xor(smax, 16));             // other 8 key rows, same query
    if (lane < 16) tmax[kt][qt * 16 + lane] = smax;
    __syncthreads();

    if (tid < 32) {                                       // combine 4 key sub-tiles + running max
      const float mp = m_state[tid];
      float mn = fmaxf(fmaxf(tmax[0][tid], tmax[1][tid]), fmaxf(tmax[2][tid], tmax[3][tid]));
      mn = fmaxf(mn, mp);
      m_new_s[tid] = mn;
      cfac_s[tid] = exp2f(mp - mn);
      m_state[tid] = mn;
    }
    __syncthreads();

    {
      const float mn = m_new_s[qt * 16 + cq];
      float psum = 0.f;
      union { bf16_t h[8]; float4 q; } up;
#pragma unroll
      for (int e = 0; e < 8; ++e) {
        const float p = exp2f(s[e] - mn);
        psum += p;
        up.h[e] = (bf16_t)p;
      }
      // lane-aligned publish: consumer lane l's B-frag bytes == producer lane l's components
      pfragq[kp][qt][lane][khalf] = up.q;
      psum += __shfl_xor(psum, 16);
      if (lane < 16) tsum[kt][qt * 16 + lane] = psum;
    }
    __syncthreads();

    if (tid < 32)
      l_state[tid] = l_state[tid] * cfac_s[tid]
                   + tsum[0][tid] + tsum[1][tid] + tsum[2][tid] + tsum[3][tid];

    // ---- phase 2: rescale + O^T[d-slice] += V^T @ P^T
    const float cf0 = cfac_s[cq];
    const float cf1 = cfac_s[16 + cq];
#pragma unroll
    for (int dt = 0; dt < 8; ++dt)
#pragma unroll
      for (int e = 0; e < 8; ++e) { oacc[dt][0][e] *= cf0; oacc[dt][1][e] *= cf1; }

#pragma unroll
    for (int kk = 0; kk < 2; ++kk) {
      FragU p0, p1;
      p0.q[0] = pfragq[kk][0][lane][0]; p0.q[1] = pfragq[kk][0][lane][1];
      p1.q[0] = pfragq[kk][1][lane][0]; p1.q[1] = pfragq[kk][1][lane][1];
      const int jj = j0 + kk * 32;
#pragma unroll
      for (int dt = 0; dt < 8; ++dt) {
        bf16x16 av = load_frag(VTb, d0 + dt * 16, NN, jj);  // A = V^T rows (contiguous over keys)
        oacc[dt][0] = wmma_bf16(av, p0.v, oacc[dt][0]);
        oacc[dt][1] = wmma_bf16(av, p1.v, oacc[dt][1]);
      }
    }
    __syncthreads();  // protect pfrag/tmax/tsum before next iteration
  }

  // normalize and store: O^T lane layout -> 8 contiguous d per accumulator -> 2x float4 stores
  const float inv0 = 1.0f / l_state[cq];
  const float inv1 = 1.0f / l_state[16 + cq];
  const int rh = (lane >> 4) << 3;
#pragma unroll
  for (int dt = 0; dt < 8; ++dt) {
#pragma unroll
    for (int q2 = 0; q2 < 2; ++q2) {
      const int qrow = qBase + q2 * 16 + cq;
      const int dcol = d0 + dt * 16 + rh;
      const float inv = q2 ? inv1 : inv0;
      float4 lo, hi;
      lo.x = oacc[dt][q2][0] * inv; lo.y = oacc[dt][q2][1] * inv;
      lo.z = oacc[dt][q2][2] * inv; lo.w = oacc[dt][q2][3] * inv;
      hi.x = oacc[dt][q2][4] * inv; hi.y = oacc[dt][q2][5] * inv;
      hi.z = oacc[dt][q2][6] * inv; hi.w = oacc[dt][q2][7] * inv;
      float* dst = out + (size_t)qrow * OO + dcol;
      *reinterpret_cast<float4*>(dst) = lo;
      *reinterpret_cast<float4*>(dst + 4) = hi;
    }
  }
}

// ---------------------------------------------------------------- launch
extern "C" void kernel_launch(void* const* d_in, const int* in_sizes, int n_in,
                              void* d_out, int out_size, void* d_ws, size_t ws_size,
                              hipStream_t stream) {
  const float* x  = (const float*)d_in[0];
  const float* Wq = (const float*)d_in[1];
  const float* bq = (const float*)d_in[2];
  const float* Wk = (const float*)d_in[3];
  const float* bk = (const float*)d_in[4];
  const float* Wv = (const float*)d_in[5];
  const float* bv = (const float*)d_in[6];
  float* out = (float*)d_out;

  // workspace layout (bf16): x, Wq, Wk, Wv, Q, K, V^T  -> ~70 MB total
  const size_t szX = (size_t)NN * HH * sizeof(bf16_t);
  const size_t szW = (size_t)OO * HH * sizeof(bf16_t);
  const size_t szM = (size_t)NN * OO * sizeof(bf16_t);
  if (ws_size < szX + 3 * szW + 3 * szM) return;

  char* ws = (char*)d_ws;
  bf16_t* Xb  = (bf16_t*)ws;  ws += szX;
  bf16_t* Wqb = (bf16_t*)ws;  ws += szW;
  bf16_t* Wkb = (bf16_t*)ws;  ws += szW;
  bf16_t* Wvb = (bf16_t*)ws;  ws += szW;
  bf16_t* Qb  = (bf16_t*)ws;  ws += szM;
  bf16_t* Kb  = (bf16_t*)ws;  ws += szM;
  bf16_t* VTb = (bf16_t*)ws;  ws += szM;

  f32_to_bf16_kernel<<<(NN * HH / 4 + 255) / 256, 256, 0, stream>>>(x,  Xb,  NN * HH / 4);
  f32_to_bf16_kernel<<<(OO * HH / 4 + 255) / 256, 256, 0, stream>>>(Wq, Wqb, OO * HH / 4);
  f32_to_bf16_kernel<<<(OO * HH / 4 + 255) / 256, 256, 0, stream>>>(Wk, Wkb, OO * HH / 4);
  f32_to_bf16_kernel<<<(OO * HH / 4 + 255) / 256, 256, 0, stream>>>(Wv, Wvb, OO * HH / 4);

  qkv_kernel<<<dim3(NN / 128, OO / 128, 3), 256, 0, stream>>>(
      Xb, Wqb, Wkb, Wvb, bq, bk, bv, Qb, Kb, VTb);

  attn_kernel<<<dim3(NN / 32), 256, 0, stream>>>(Qb, Kb, VTb, out);
}
